// SAE_3831110828649
// MI455X (gfx1250) — matrix operand order repaired and checked by reading, not compiled, not run on hardware
//
#include <hip/hip_runtime.h>
#include <stdint.h>

typedef __attribute__((ext_vector_type(16))) _Float16 v16h;
typedef __attribute__((ext_vector_type(8)))  float    v8f;

#define MODEL_K   2048              // model dim (K of encode GEMM, cols of decoder)
#define HIDDEN_H  16384             // hidden dim
#define N_TOKENS  16384
#define ROWS      32                // tokens per workgroup (2 strips of 16)
#define WAVES     8
#define TPB       256
#define KSTEPS    (MODEL_K / 32)    // 64 k-steps of 32
#define CTILES    (HIDDEN_H / 16)   // 1024 column tiles
#define TKK       64                // top-k

// ---------- helpers ----------
__device__ __forceinline__ unsigned short f2h(float f) {
  return __builtin_bit_cast(unsigned short, (_Float16)f);
}
// order-preserving float -> u32 key (monotone increasing)
__device__ __forceinline__ unsigned skey(float f) {
  unsigned u = __builtin_bit_cast(unsigned, f);
  return (u & 0x80000000u) ? ~u : (u | 0x80000000u);
}
__device__ __forceinline__ float unskey(unsigned k) {
  unsigned u = (k & 0x80000000u) ? (k ^ 0x80000000u) : ~k;
  return __builtin_bit_cast(float, u);
}
#define TG_INIT 0x00800000u   // skey(-FLT_MAX): "no threshold yet"

__device__ __forceinline__ unsigned wave_min_u32(unsigned v, int lane) {
#pragma unroll
  for (int off = 16; off >= 1; off >>= 1) {
    unsigned o = (unsigned)__builtin_amdgcn_ds_bpermute(((lane ^ off) << 2), (int)v);
    v = (o < v) ? o : v;
  }
  return v;
}
__device__ __forceinline__ unsigned long long wave_max_u64(unsigned long long v, int lane) {
#pragma unroll
  for (int off = 16; off >= 1; off >>= 1) {
    int idx = ((lane ^ off) << 2);
    unsigned lo  = (unsigned)v, hi = (unsigned)(v >> 32);
    unsigned olo = (unsigned)__builtin_amdgcn_ds_bpermute(idx, (int)lo);
    unsigned ohi = (unsigned)__builtin_amdgcn_ds_bpermute(idx, (int)hi);
    unsigned long long o = ((unsigned long long)ohi << 32) | olo;
    v = (o > v) ? o : v;
  }
  return v;
}

// exact min (value-key and position) of a full 64-entry value list (2 entries/lane)
__device__ __forceinline__ void list_min(const float* Lv, int lane, unsigned& kmin, int& pos) {
  float2 q = *(const float2*)(Lv + 2 * lane);
  unsigned k0 = skey(q.x), k1 = skey(q.y);
  int      slot = (k1 < k0) ? 1 : 0;
  unsigned lk   = slot ? k1 : k0;
  kmin = wave_min_u32(lk, lane);
  unsigned bal = __builtin_amdgcn_ballot_w32(lk == kmin);
  int L  = (int)__builtin_ctz(bal);
  int sl = __builtin_amdgcn_readlane(slot, L);
  pos = 2 * L + sl;
}

// whole-wave cooperative insert of uniform (v,col) into this wave-row's list.
// Exactness: a candidate may be dropped only if 64 kept values exceed it
// (list replace-min), and Tg = max over waves of list-mins <= true 64th value.
__device__ __forceinline__ void insert_cand(float* Lv, unsigned short* Lc, int* cntp,
                                            unsigned* tgp, int lane, float v, int col) {
  int c = *cntp;
  if (c < TKK) {                       // fill phase: append
    if (lane == 0) { Lv[c] = v; Lc[c] = (unsigned short)col; *cntp = c + 1; }
    if (c + 1 == TKK) {                // seed global threshold
      unsigned kmin; int pos;
      list_min(Lv, lane, kmin, pos);
      if (lane == 0) atomicMax(tgp, kmin);
    }
  } else {
    if (skey(v) <= *tgp) return;       // fresh global-threshold re-check
    unsigned kmin; int pos;
    list_min(Lv, lane, kmin, pos);
    if (skey(v) > kmin) {
      if (lane == 0) { Lv[pos] = v; Lc[pos] = (unsigned short)col; }
    }
    if (lane == 0) atomicMax(tgp, kmin);  // push (lagged) list-min into Tg
  }
}

// ---------- pre-pass: encoder f32 [K,H] -> f16 B-fragment panels in ws ----------
// frag(t,s): 32 lanes x 32B. lane l: col n = t*16 + (l&15), K = s*32 + (l>>4)*16 + i
__global__ void __launch_bounds__(TPB) convert_encoder(const float* __restrict__ enc,
                                                       unsigned short* __restrict__ bws) {
  int tid  = blockIdx.x * TPB + threadIdx.x;        // 0 .. 2M-1
  int lane = tid & 31;
  int s    = (tid >> 5) & (KSTEPS - 1);
  int t    = tid >> 11;
  int k0   = s * 32 + ((lane >> 4) << 4);
  int hcol = t * 16 + (lane & 15);
  const float* p = enc + (size_t)k0 * HIDDEN_H + hcol;
  unsigned v[8];
#pragma unroll
  for (int i = 0; i < 8; i++) {
    unsigned short lo = f2h(p[(size_t)(2 * i) * HIDDEN_H]);
    unsigned short hi = f2h(p[(size_t)(2 * i + 1) * HIDDEN_H]);
    v[i] = (unsigned)lo | ((unsigned)hi << 16);
  }
  uint4* d = (uint4*)(bws + (size_t)tid * 16);
  d[0] = make_uint4(v[0], v[1], v[2], v[3]);
  d[1] = make_uint4(v[4], v[5], v[6], v[7]);
}

// ---------- fused encode-GEMM + exact top-64 + sparse decode ----------
__global__ void __launch_bounds__(TPB) sae_fused(const float* __restrict__ x,
                                                 const float* __restrict__ encb,
                                                 const float* __restrict__ dec,
                                                 const float* __restrict__ decb,
                                                 const unsigned short* __restrict__ bws,
                                                 float* __restrict__ out) {
  // A fragments: [s][strip][lane][16 halves]  (32 tokens x 2048 K, f16) = 128 KB
  __shared__ __align__(32) unsigned short As[KSTEPS * 2 * 32 * 16];
  __shared__ __align__(16) float          listV[WAVES * ROWS * TKK];   // 64 KB
  __shared__ __align__(16) unsigned short listC[WAVES * ROWS * TKK];   // 32 KB
  __shared__ __align__(16) unsigned int   finalL[ROWS * TKK * 2];      // 16 KB
  __shared__ unsigned TgKey[ROWS];            // shared per-row threshold (sortable key)
  __shared__ int      cntSh[WAVES * ROWS];

  const int tid  = threadIdx.x;
  const int lane = tid & 31;
  const int wv   = tid >> 5;
  const int row0 = blockIdx.x * ROWS;

  // ---- stage x tile into LDS as f16 A-fragments (both strips) ----
  // unit u: run=u&1, ln=(u>>1)&31, strip=(u>>6)&1, s=u>>7
  for (int u = tid; u < KSTEPS * 2 * 32 * 2; u += TPB) {
    int run = u & 1;
    int ln  = (u >> 1) & 31;
    int ss  = (u >> 6) & 1;
    int s   = u >> 7;
    int m   = (ln & 15) + (ss << 4);
    int kst = s * 32 + ((ln >> 4) << 4) + run * 8;
    const float4* src = (const float4*)(x + (size_t)(row0 + m) * MODEL_K + kst);
    float4 a = src[0];
    float4 b = src[1];
    unsigned v0 = (unsigned)f2h(a.x) | ((unsigned)f2h(a.y) << 16);
    unsigned v1 = (unsigned)f2h(a.z) | ((unsigned)f2h(a.w) << 16);
    unsigned v2 = (unsigned)f2h(b.x) | ((unsigned)f2h(b.y) << 16);
    unsigned v3 = (unsigned)f2h(b.z) | ((unsigned)f2h(b.w) << 16);
    *(uint4*)(As + (((s * 2 + ss) * 32 + ln) * 16 + run * 8)) = make_uint4(v0, v1, v2, v3);
  }
  for (int i = tid; i < WAVES * ROWS * TKK; i += TPB) listV[i] = -3.402823466e38f;
  if (tid < WAVES * ROWS) cntSh[tid] = 0;
  if (tid < ROWS) TgKey[tid] = TG_INIT;
  __syncthreads();

  // ---- encode GEMM: each wave does 128 column tiles x BOTH 16-row strips ----
  for (int t = wv; t < CTILES; t += WAVES) {
    const char* bp = (const char*)bws + ((size_t)t * KSTEPS * 32 + lane) * 32;
    const char* ap = (const char*)As + lane * 32;
    v8f aE0 = {}, aE1 = {}, aO0 = {}, aO1 = {};
#pragma unroll 4
    for (int s = 0; s < KSTEPS; s += 2) {
      v16h a00 = *(const v16h*)(ap);            // s even, strip 0
      v16h a01 = *(const v16h*)(ap + 1024);     // s even, strip 1
      v16h a10 = *(const v16h*)(ap + 2048);     // s odd,  strip 0
      v16h a11 = *(const v16h*)(ap + 3072);     // s odd,  strip 1
      v16h b0  = *(const v16h*)(bp);
      v16h b1  = *(const v16h*)(bp + 1024);
      aE0 = __builtin_amdgcn_wmma_f32_16x16x32_f16(false, a00, false, b0, (short)0, aE0, false, false);
      aE1 = __builtin_amdgcn_wmma_f32_16x16x32_f16(false, a01, false, b0, (short)0, aE1, false, false);
      aO0 = __builtin_amdgcn_wmma_f32_16x16x32_f16(false, a10, false, b1, (short)0, aO0, false, false);
      aO1 = __builtin_amdgcn_wmma_f32_16x16x32_f16(false, a11, false, b1, (short)0, aO1, false, false);
      ap += 4096;
      bp += 2048;
    }
    float bias = encb[t * 16 + (lane & 15)];
    v8f accL, accH;                       // strip 0 (rows 0-15), strip 1 (rows 16-31)
#pragma unroll
    for (int r = 0; r < 8; r++) {
      accL[r] = aE0[r] + aO0[r] + bias;
      accH[r] = aE1[r] + aO1[r] + bias;
    }

    // C layout: reg r -> row = r + 8*(lane>=16) (+16 for strip1), col = t*16+(lane&15)
    const int hb = lane >> 4;
#pragma unroll
    for (int r = 0; r < 8; r++) {
      float t0 = unskey(TgKey[r + (hb << 3)]);
      float t1 = unskey(TgKey[16 + r + (hb << 3)]);
      unsigned m0 = __builtin_amdgcn_ballot_w32(accL[r] > t0);
      unsigned m1 = __builtin_amdgcn_ballot_w32(accH[r] > t1);
      while (m0) {
        int src = __builtin_ctz(m0);
        m0 &= m0 - 1;
        float v = __builtin_bit_cast(float,
                    __builtin_amdgcn_readlane(__builtin_bit_cast(int, accL[r]), src));
        int row = r + ((src >> 4) << 3);
        int col = t * 16 + (src & 15);
        int wr  = wv * ROWS + row;
        insert_cand(listV + (size_t)wr * TKK, listC + (size_t)wr * TKK,
                    &cntSh[wr], &TgKey[row], lane, v, col);
      }
      while (m1) {
        int src = __builtin_ctz(m1);
        m1 &= m1 - 1;
        float v = __builtin_bit_cast(float,
                    __builtin_amdgcn_readlane(__builtin_bit_cast(int, accH[r]), src));
        int row = 16 + r + ((src >> 4) << 3);
        int col = t * 16 + (src & 15);
        int wr  = wv * ROWS + row;
        insert_cand(listV + (size_t)wr * TKK, listC + (size_t)wr * TKK,
                    &cntSh[wr], &TgKey[row], lane, v, col);
      }
    }
  }
  __syncthreads();

  // ---- merge: exact top-64 of the 8x64 per-row candidates ----
  for (int row = wv; row < ROWS; row += WAVES) {
    unsigned long long kk[16];
#pragma unroll
    for (int i = 0; i < 16; i++) {
      int c = lane + 32 * i;                      // 512 candidates
      int idx = ((c >> 6) * ROWS + row) * TKK + (c & 63);
      float    vv = listV[idx];
      unsigned cl = listC[idx];
      kk[i] = ((unsigned long long)skey(vv) << 32) | cl;   // cols unique across waves
    }
    for (int j = 0; j < TKK; j++) {
      unsigned long long best = kk[0];
#pragma unroll
      for (int i = 1; i < 16; i++) best = (kk[i] > best) ? kk[i] : best;
      best = wave_max_u64(best, lane);
#pragma unroll
      for (int i = 0; i < 16; i++)
        if (kk[i] == best) kk[i] = 0;
      if (lane == 0) {
        finalL[(row * TKK + j) * 2]     = __builtin_bit_cast(unsigned, unskey((unsigned)(best >> 32)));
        finalL[(row * TKK + j) * 2 + 1] = (unsigned)(best & 0xffffu);
      }
    }
  }
  __syncthreads();

  // ---- sparse decode: out[row,:] = decb + sum relu(val)*dec[col,:] ----
  for (int row = wv; row < ROWS; row += WAVES) {
    float4 acc[16];
#pragma unroll
    for (int i = 0; i < 16; i++) acc[i] = ((const float4*)decb)[lane + 32 * i];
    for (int j = 0; j < TKK; j++) {
      float    v  = __builtin_bit_cast(float, finalL[(row * TKK + j) * 2]);
      unsigned cl = finalL[(row * TKK + j) * 2 + 1];
      if (v > 0.0f) {                    // relu: non-positive entries contribute nothing
        const float4* dr = (const float4*)(dec + (size_t)cl * MODEL_K);
#pragma unroll
        for (int i = 0; i < 16; i++) {
          float4 d4 = dr[lane + 32 * i];
          acc[i].x += v * d4.x;
          acc[i].y += v * d4.y;
          acc[i].z += v * d4.z;
          acc[i].w += v * d4.w;
        }
      }
    }
    float4* op = (float4*)(out + (size_t)(row0 + row) * MODEL_K);
#pragma unroll
    for (int i = 0; i < 16; i++) op[lane + 32 * i] = acc[i];
  }
}

extern "C" void kernel_launch(void* const* d_in, const int* in_sizes, int n_in,
                              void* d_out, int out_size, void* d_ws, size_t ws_size,
                              hipStream_t stream) {
  (void)in_sizes; (void)n_in; (void)out_size; (void)ws_size;
  const float* x    = (const float*)d_in[0];
  const float* enc  = (const float*)d_in[1];
  const float* encb = (const float*)d_in[2];
  const float* dec  = (const float*)d_in[3];
  const float* decb = (const float*)d_in[4];
  float* out = (float*)d_out;
  unsigned short* bws = (unsigned short*)d_ws;   // 64 MB f16 swizzled encoder panels

  convert_encoder<<<(CTILES * KSTEPS * 32) / TPB, TPB, 0, stream>>>(enc, bws);
  sae_fused<<<N_TOKENS / ROWS, TPB, 0, stream>>>(x, encb, dec, decb, bws, out);
}
